// ScanModule_5978594476775
// MI455X (gfx1250) — matrix operand order repaired and checked by reading, not compiled, not run on hardware
//
#include <hip/hip_runtime.h>
#include <hip/hip_bf16.h>
#include <math.h>

// ---------------------------------------------------------------------------
// CDNA5 (gfx1250) RNN scan:  h' = tanh(x_t@Wx + h@Wh); y_t = h'@Wo
// fp32 accuracy via 3xBF16 split feeding v_wmma_f32_16x16x32_bf16 (wave32).
// Recurrence runs in ONE persistent kernel (16 blocks) with a device-scope
// atomic barrier between steps; h_t is stored pre-split (bf16 hi/lo) so the
// next step and the final H@Wo GEMM consume WMMA A-fragments directly.
// ---------------------------------------------------------------------------

typedef __attribute__((ext_vector_type(16))) __bf16 v16bf;
typedef __attribute__((ext_vector_type(8)))  float  v8f;

union BF16Frag {
  v16bf v;
  uint4 q[2];   // 2 x 16B chunks = 16 bf16
};

__device__ __forceinline__ unsigned short f32_to_bf16_rne(float f) {
  unsigned u = __builtin_bit_cast(unsigned, f);
  unsigned r = u + 0x7FFFu + ((u >> 16) & 1u);   // round-to-nearest-even
  return (unsigned short)(r >> 16);
}
__device__ __forceinline__ float bf16_to_f32(unsigned short h) {
  unsigned u = ((unsigned)h) << 16;
  return __builtin_bit_cast(float, u);
}

__device__ __forceinline__ v8f wmma_bf16(const BF16Frag& a, const BF16Frag& b, v8f c) {
  return __builtin_amdgcn_wmma_f32_16x16x32_bf16(false, a.v, false, b.v,
                                                 (short)0, c, false, false);
}

// ---------------------------------------------------------------------------
// Weight prep: W[K][N] fp32 -> Wt_hi/Wt_lo[N][K] bf16 split (B-operand layout:
// each lane's WMMA B-fragment = two 8-K runs (k0+8h, k0+16+8h) of one column = 2 b128 loads).
// ---------------------------------------------------------------------------
__global__ void wsplitT_kernel(const float* __restrict__ W,
                               unsigned short* __restrict__ Wt_hi,
                               unsigned short* __restrict__ Wt_lo,
                               int K, int N) {
  int idx = blockIdx.x * blockDim.x + threadIdx.x;
  int k = idx / N;
  int n = idx % N;
  float f = W[(size_t)k * N + n];
  unsigned short hi = f32_to_bf16_rne(f);
  unsigned short lo = f32_to_bf16_rne(f - bf16_to_f32(hi));
  Wt_hi[(size_t)n * K + k] = hi;
  Wt_lo[(size_t)n * K + k] = lo;
}

// Row-major fp32 -> bf16 hi/lo split (for h0; layout matches A-operand use).
__global__ void split_rows_kernel(const float* __restrict__ A,
                                  unsigned short* __restrict__ A_hi,
                                  unsigned short* __restrict__ A_lo) {
  int idx = blockIdx.x * blockDim.x + threadIdx.x;
  float f = A[idx];
  unsigned short hi = f32_to_bf16_rne(f);
  unsigned short lo = f32_to_bf16_rne(f - bf16_to_f32(hi));
  A_hi[idx] = hi;
  A_lo[idx] = lo;
}

// ---------------------------------------------------------------------------
// Variant 1: fp32 A staged+split through LDS (conversion amortized 8x across
// the block's waves). Used for Xp = x @ Wx. Block = 8 wave32's, tile 64x128.
// ---------------------------------------------------------------------------
template <int MT>
__global__ void gemm_stageA_kernel(const float* __restrict__ A,             // [M,K] f32
                                   const unsigned short* __restrict__ Bt_hi,// [N,K]
                                   const unsigned short* __restrict__ Bt_lo,
                                   float* __restrict__ C,                   // [M,N]
                                   int M, int N, int K) {
  constexpr int BM = MT * 16;
  constexpr int KB = 32;
  __shared__ unsigned short sA[2 * BM * KB];
  unsigned short* sA_hi = sA;
  unsigned short* sA_lo = sA + BM * KB;

  const int tid  = threadIdx.x;
  const int lane = tid & 31;
  const int wave = tid >> 5;
  const int blocks_n = N >> 7;
  const int bm   = blockIdx.x / blocks_n;
  const int bn   = blockIdx.x % blocks_n;
  const int row0 = bm * BM;
  const int ncol = bn * 128 + wave * 16 + (lane & 15);
  const int kgrp = lane >> 4;

  v8f acc[MT];
  v8f zero = {0.f, 0.f, 0.f, 0.f, 0.f, 0.f, 0.f, 0.f};
#pragma unroll
  for (int s = 0; s < MT; ++s) acc[s] = zero;

  const unsigned short* bph = Bt_hi + (size_t)ncol * K;
  const unsigned short* bpl = Bt_lo + (size_t)ncol * K;

  for (int k0 = 0; k0 < K; k0 += KB) {
    __syncthreads();
#pragma unroll
    for (int i = 0; i < (BM * KB) / 256; ++i) {
      int e  = i * 256 + tid;
      int r  = e >> 5;
      int kk = e & 31;
      float f = A[(size_t)(row0 + r) * K + (k0 + kk)];
      unsigned short hi = f32_to_bf16_rne(f);
      unsigned short lo = f32_to_bf16_rne(f - bf16_to_f32(hi));
      sA_hi[r * KB + kk] = hi;
      sA_lo[r * KB + kk] = lo;
    }
    __syncthreads();

    const int kb = k0 + kgrp * 8;
    BF16Frag bh, bl;
    bh.q[0] = *(const uint4*)(bph + kb);
    bh.q[1] = *(const uint4*)(bph + kb + 16);
    bl.q[0] = *(const uint4*)(bpl + kb);
    bl.q[1] = *(const uint4*)(bpl + kb + 16);
    __builtin_prefetch(bph + kb + KB, 0, 1);
    __builtin_prefetch(bpl + kb + KB, 0, 1);

#pragma unroll
    for (int s = 0; s < MT; ++s) {
      const unsigned short* ah = &sA_hi[(s * 16 + (lane & 15)) * KB + kgrp * 8];
      const unsigned short* al = &sA_lo[(s * 16 + (lane & 15)) * KB + kgrp * 8];
      BF16Frag fah, fal;
      fah.q[0] = *(const uint4*)(ah);
      fah.q[1] = *(const uint4*)(ah + 16);
      fal.q[0] = *(const uint4*)(al);
      fal.q[1] = *(const uint4*)(al + 16);
      acc[s] = wmma_bf16(fah, bh, acc[s]);
      acc[s] = wmma_bf16(fah, bl, acc[s]);
      acc[s] = wmma_bf16(fal, bh, acc[s]);
    }
  }

#pragma unroll
  for (int s = 0; s < MT; ++s) {
#pragma unroll
    for (int r = 0; r < 8; ++r) {
      int m = row0 + s * 16 + 8 * kgrp + r;   // C layout: M = r + 8*(lane>>4)
      C[(size_t)m * N + ncol] = acc[s][r];
    }
  }
}

// ---------------------------------------------------------------------------
// Variant 2: pre-split bf16 A consumed directly from global (no LDS, no
// barriers -> fully pipelineable K loop). Used for ys = H @ Wo.
// ---------------------------------------------------------------------------
template <int MT>
__global__ void gemm_directA_kernel(const unsigned short* __restrict__ A_hi, // [M,K]
                                    const unsigned short* __restrict__ A_lo,
                                    const unsigned short* __restrict__ Bt_hi,// [N,K]
                                    const unsigned short* __restrict__ Bt_lo,
                                    float* __restrict__ C,                   // [M,N]
                                    int M, int N, int K) {
  const int tid  = threadIdx.x;
  const int lane = tid & 31;
  const int wave = tid >> 5;
  const int blocks_n = N >> 7;
  const int bm   = blockIdx.x / blocks_n;
  const int bn   = blockIdx.x % blocks_n;
  const int row0 = bm * MT * 16;
  const int ncol = bn * 128 + wave * 16 + (lane & 15);
  const int kgrp = lane >> 4;
  const int arow = lane & 15;

  v8f acc[MT];
  v8f zero = {0.f, 0.f, 0.f, 0.f, 0.f, 0.f, 0.f, 0.f};
#pragma unroll
  for (int s = 0; s < MT; ++s) acc[s] = zero;

  const unsigned short* bph = Bt_hi + (size_t)ncol * K;
  const unsigned short* bpl = Bt_lo + (size_t)ncol * K;

  for (int k0 = 0; k0 < K; k0 += 32) {
    const int kb = k0 + kgrp * 8;
    BF16Frag bh, bl;
    bh.q[0] = *(const uint4*)(bph + kb);
    bh.q[1] = *(const uint4*)(bph + kb + 16);
    bl.q[0] = *(const uint4*)(bpl + kb);
    bl.q[1] = *(const uint4*)(bpl + kb + 16);
    __builtin_prefetch(bph + kb + 32, 0, 1);
    __builtin_prefetch(bpl + kb + 32, 0, 1);

#pragma unroll
    for (int s = 0; s < MT; ++s) {
      const size_t aoff = (size_t)(row0 + s * 16 + arow) * K + k0 + kgrp * 8;
      BF16Frag fah, fal;
      fah.q[0] = *(const uint4*)(A_hi + aoff);
      fah.q[1] = *(const uint4*)(A_hi + aoff + 16);
      fal.q[0] = *(const uint4*)(A_lo + aoff);
      fal.q[1] = *(const uint4*)(A_lo + aoff + 16);
      acc[s] = wmma_bf16(fah, bh, acc[s]);
      acc[s] = wmma_bf16(fah, bl, acc[s]);
      acc[s] = wmma_bf16(fal, bh, acc[s]);
    }
  }

#pragma unroll
  for (int s = 0; s < MT; ++s) {
#pragma unroll
    for (int r = 0; r < 8; ++r) {
      int m = row0 + s * 16 + 8 * kgrp + r;
      C[(size_t)m * N + ncol] = acc[s][r];
    }
  }
}

// ---------------------------------------------------------------------------
// Persistent recurrence kernel: 16 blocks x 8 waves = 128 waves, one 16-col
// strip each. All T steps internal; steps separated by a device-scope atomic
// barrier. Epilogue fuses tanh + bf16 hi/lo re-split of h_t.
// ---------------------------------------------------------------------------
__global__ void scan_kernel(const unsigned short* __restrict__ h0_hi,
                            const unsigned short* __restrict__ h0_lo,
                            const unsigned short* __restrict__ WhT_hi, // [D,D]
                            const unsigned short* __restrict__ WhT_lo,
                            const float* __restrict__ Xp,              // [T,B,D]
                            unsigned short* __restrict__ Hs_hi,        // [T*B,D]
                            unsigned short* __restrict__ Hs_lo,
                            float* __restrict__ h_final,               // [B,D]
                            unsigned* __restrict__ bar,
                            int T) {
  constexpr int D  = 2048;
  constexpr int NB = 16;                       // batch rows
  const int tid  = threadIdx.x;
  const int lane = tid & 31;
  const int wave = tid >> 5;
  const int ncol = blockIdx.x * 128 + wave * 16 + (lane & 15);
  const int kgrp = lane >> 4;
  const int arow = lane & 15;
  const unsigned nblocks = gridDim.x;

  const unsigned short* bph = WhT_hi + (size_t)ncol * D;
  const unsigned short* bpl = WhT_lo + (size_t)ncol * D;
  v8f zero = {0.f, 0.f, 0.f, 0.f, 0.f, 0.f, 0.f, 0.f};

  for (int t = 0; t < T; ++t) {
    const unsigned short* ah_base = (t == 0) ? h0_hi : Hs_hi + (size_t)(t - 1) * NB * D;
    const unsigned short* al_base = (t == 0) ? h0_lo : Hs_lo + (size_t)(t - 1) * NB * D;
    const unsigned short* arh = ah_base + (size_t)arow * D + kgrp * 8;
    const unsigned short* arl = al_base + (size_t)arow * D + kgrp * 8;

    v8f acc = zero;
#pragma unroll 2
    for (int k0 = 0; k0 < D; k0 += 32) {
      BF16Frag fah, fal, bh, bl;
      fah.q[0] = *(const uint4*)(arh + k0);
      fah.q[1] = *(const uint4*)(arh + k0 + 16);
      fal.q[0] = *(const uint4*)(arl + k0);
      fal.q[1] = *(const uint4*)(arl + k0 + 16);
      const int kb = k0 + kgrp * 8;
      bh.q[0] = *(const uint4*)(bph + kb);
      bh.q[1] = *(const uint4*)(bph + kb + 16);
      bl.q[0] = *(const uint4*)(bpl + kb);
      bl.q[1] = *(const uint4*)(bpl + kb + 16);
      __builtin_prefetch(bph + kb + 32, 0, 1);
      __builtin_prefetch(bpl + kb + 32, 0, 1);
      acc = wmma_bf16(fah, bh, acc);
      acc = wmma_bf16(fah, bl, acc);
      acc = wmma_bf16(fal, bh, acc);
    }

    // epilogue: bias(Xp[t]) + tanh, write split h_t (+ fp32 h_final at last t)
    unsigned short* oh = Hs_hi + (size_t)t * NB * D;
    unsigned short* ol = Hs_lo + (size_t)t * NB * D;
    const float* bias = Xp + (size_t)t * NB * D;
#pragma unroll
    for (int r = 0; r < 8; ++r) {
      int m = 8 * kgrp + r;
      size_t off = (size_t)m * D + ncol;
      float v = tanhf(acc[r] + bias[off]);
      unsigned short hi = f32_to_bf16_rne(v);
      unsigned short lo = f32_to_bf16_rne(v - bf16_to_f32(hi));
      oh[off] = hi;
      ol[off] = lo;
      if (t == T - 1) h_final[off] = v;
    }

    // device-scope barrier between steps (release writes, acquire reads)
    __syncthreads();
    if (tid == 0) {
      __threadfence();
      __hip_atomic_fetch_add(bar, 1u, __ATOMIC_RELEASE, __HIP_MEMORY_SCOPE_AGENT);
      const unsigned target = nblocks * (unsigned)(t + 1);
      while (__hip_atomic_load(bar, __ATOMIC_ACQUIRE, __HIP_MEMORY_SCOPE_AGENT) < target) {
        __builtin_amdgcn_s_sleep(1);
      }
      __threadfence();
    }
    __syncthreads();
  }
}

// ---------------------------------------------------------------------------
extern "C" void kernel_launch(void* const* d_in, const int* in_sizes, int n_in,
                              void* d_out, int out_size, void* d_ws, size_t ws_size,
                              hipStream_t stream) {
  (void)in_sizes; (void)n_in; (void)out_size; (void)ws_size;
  const int T = 512, B = 16, D = 2048;

  const float* h0 = (const float*)d_in[0];
  const float* x  = (const float*)d_in[1];   // [T,B,D]
  const float* Wx = (const float*)d_in[2];
  const float* Wh = (const float*)d_in[3];
  const float* Wo = (const float*)d_in[4];

  float* out_h = (float*)d_out;              // [B,D]
  float* out_y = out_h + (size_t)B * D;      // [T,B,D]

  // ---- workspace carve-up (~176 MB) ----
  char* p = (char*)d_ws;
  const size_t wbytes = (size_t)D * D * sizeof(unsigned short);
  unsigned short* WxT_hi = (unsigned short*)p; p += wbytes;
  unsigned short* WxT_lo = (unsigned short*)p; p += wbytes;
  unsigned short* WhT_hi = (unsigned short*)p; p += wbytes;
  unsigned short* WhT_lo = (unsigned short*)p; p += wbytes;
  unsigned short* WoT_hi = (unsigned short*)p; p += wbytes;
  unsigned short* WoT_lo = (unsigned short*)p; p += wbytes;
  float*          Xp     = (float*)p;          p += (size_t)T * B * D * sizeof(float);
  unsigned short* Hs_hi  = (unsigned short*)p; p += (size_t)T * B * D * sizeof(unsigned short);
  unsigned short* Hs_lo  = (unsigned short*)p; p += (size_t)T * B * D * sizeof(unsigned short);
  unsigned short* h0_hi  = (unsigned short*)p; p += (size_t)B * D * sizeof(unsigned short);
  unsigned short* h0_lo  = (unsigned short*)p; p += (size_t)B * D * sizeof(unsigned short);
  unsigned*       bar    = (unsigned*)p;       p += 256;

  // ---- per-call init of barrier counter (no cross-call state) ----
  hipMemsetAsync(bar, 0, sizeof(unsigned), stream);

  // ---- one-time weight transpose + bf16 hi/lo split (stays L2-resident) ----
  const int wblocks = (D * D) / 256;
  wsplitT_kernel<<<wblocks, 256, 0, stream>>>(Wx, WxT_hi, WxT_lo, D, D);
  wsplitT_kernel<<<wblocks, 256, 0, stream>>>(Wh, WhT_hi, WhT_lo, D, D);
  wsplitT_kernel<<<wblocks, 256, 0, stream>>>(Wo, WoT_hi, WoT_lo, D, D);
  split_rows_kernel<<<(B * D) / 256, 256, 0, stream>>>(h0, h0_hi, h0_lo);

  // ---- parallel precompute: Xp = x @ Wx  (M = 8192) ----
  {
    const int M = T * B;
    gemm_stageA_kernel<4><<<(M / 64) * (D / 128), 256, 0, stream>>>(
        x, WxT_hi, WxT_lo, Xp, M, D, D);
  }

  // ---- persistent recurrence: all 512 steps, one launch ----
  scan_kernel<<<D / 128, 256, 0, stream>>>(h0_hi, h0_lo, WhT_hi, WhT_lo, Xp,
                                           Hs_hi, Hs_lo, out_h, bar, T);

  // ---- parallel postcompute: ys = H @ Wo (A pre-split by scan epilogue) ----
  {
    const int M = T * B;
    gemm_directA_kernel<4><<<(M / 64) * (D / 128), 256, 0, stream>>>(
        Hs_hi, Hs_lo, WoT_hi, WoT_lo, out_y, M, D, D);
  }
}
